// TemporalConvBlock_26577257627924
// MI455X (gfx1250) — matrix operand order repaired and checked by reading, not compiled, not run on hardware
//
#include <hip/hip_runtime.h>

// B=16, T=512, F=32, CI=128, CO=128, K=3, DIL=2
// y[b,t,f,o] = sum_{k,i} x[b, t + k*DIL - 4, f, i] * w[f,o,i,k] + bias[f,o]
// Per feature: GEMM [M=8192, Kdim=384] x [384, N=128], Kdim ordered kk = k*CI + i.

typedef __attribute__((ext_vector_type(16))) _Float16 v16h;
typedef __attribute__((ext_vector_type(8)))  _Float16 v8h;
typedef __attribute__((ext_vector_type(4)))  _Float16 v4h;
typedef __attribute__((ext_vector_type(8)))  float    v8f;
typedef __attribute__((ext_vector_type(4)))  unsigned int u32x4;
typedef __attribute__((ext_vector_type(8)))  int      i32x8;
typedef __attribute__((ext_vector_type(4)))  int      i32x4;

#define BATCH 16
#define TLEN  512
#define NF    32
#define CI    128
#define CO    128
#define KW    3
#define DILA  2
#define KDIM  (CI*KW)        // 384
#define NKC   (KDIM/32)      // 12 K-chunks of 32
#define MTILE 128            // rows of M per workgroup
#define HALO  (DILA*(KW-1))  // 4
#define XROWS (MTILE+HALO)   // 132
#define PADROW 136           // x tile row stride in halves (272B = 68 banks, kills conflicts)
#define WSWZ_PER_F (NKC*8*512)       // 49152 f16 per feature
#define WLDS_BYTES (WSWZ_PER_F*2)    // 98304
#define XS_BYTES   (XROWS*PADROW*2)  // 35904
#define SMEM_BYTES (WLDS_BYTES + XS_BYTES)  // 134208

// ---------------------------------------------------------------------------
// Pass 1: swizzle w[f][o][i][k] (f32) into WMMA-B-fragment order (f16),
// chunked for conflict-free ds_load_b128:
//   wsw[f][kc*8+nt][chunk(=h>>3)][lane][h&7]
// B 32x16 f16 hw layout: lanes 0-15 hold K=(h&7)+16*(h>>3) for N=lane; lanes 16-31 hold
// K=8+(h&7)+16*(h>>3) for N=lane-16.
// ---------------------------------------------------------------------------
__global__ void swizzle_w_kernel(const float* __restrict__ w,
                                 _Float16* __restrict__ wsw) {
    int e = blockIdx.x * blockDim.x + threadIdx.x;   // < NF*WSWZ_PER_F
    int f    = e / WSWZ_PER_F;
    int r    = e - f * WSWZ_PER_F;
    int c    = r >> 9;           // kc*8+nt  (0..95)
    int rem  = r & 511;
    int h    = (rem >> 8) * 8 + (rem & 7);   // half index 0..15
    int lane = (rem >> 3) & 31;
    int kc   = c >> 3;
    int nt   = c & 7;
    int koff = (h & 7) + ((lane & 16) >> 1) + ((h >> 3) << 4);  // K within 32-chunk
    int kk   = kc * 32 + koff;           // kk = k*CI + i
    int o    = nt * 16 + (lane & 15);
    int i    = kk & (CI - 1);
    int k    = kk >> 7;
    wsw[e] = (_Float16)w[((f * CO + o) * CI + i) * KW + k];
}

// ---------------------------------------------------------------------------
// Main kernel: one WG = (feature f, 128-row m-block). 8 waves; each wave owns
// a 16-row strip and all 128 output columns (8 v8f accumulators).
// W_f staged into LDS via the Tensor Data Mover (raw 96KB 1-D copy of the
// pre-swizzled blob), overlapped with VALU f32->f16 staging of the x tile.
// ---------------------------------------------------------------------------
__global__ __launch_bounds__(256)
void conv_wmma_kernel(const float* __restrict__ x,
                      const _Float16* __restrict__ wsw,
                      const float* __restrict__ bias,
                      float* __restrict__ out) {
    extern __shared__ char smem[];
    _Float16* wlds = (_Float16*)smem;                 // LDS offset 0, 96 KB
    _Float16* xs   = (_Float16*)(smem + WLDS_BYTES);  // padded x tile

    int f   = blockIdx.x >> 6;            // 32 features
    int mb  = blockIdx.x & 63;            // 64 m-blocks per feature
    int bb  = mb >> 2;                    // batch index
    int t0  = (mb & 3) * MTILE;           // t range [t0, t0+128)
    int tid = threadIdx.x;
    int lane = tid & 31;
    int wv   = tid >> 5;

    // --- TDM: async DMA of pre-swizzled W_f (96 KB contiguous) into LDS ---
    if (wv == 0) {
        unsigned long long ga = (unsigned long long)(wsw + (size_t)f * WSWZ_PER_F);
        u32x4 g0; i32x8 g1;
        i32x4 g2 = {0, 0, 0, 0}, g3 = {0, 0, 0, 0};
        i32x8 g4 = {0, 0, 0, 0, 0, 0, 0, 0};
        g0[0] = 1u;                                   // count=1 (valid descriptor)
        g0[1] = 0u;                                   // lds_addr = LDS base
        g0[2] = (unsigned)(ga & 0xffffffffu);         // global_addr[31:0]
        g0[3] = (unsigned)((ga >> 32) & 0x01ffffffu)  // global_addr[56:32]
              | 0x80000000u;                          // type=2 ("image")
        g1[0] = 3 << 16;                              // data_size=8B units
        g1[1] = 0x3000 << 16;                         // tensor_dim0 = 12288
        g1[2] = 1 << 16;                              // tensor_dim1 = 1
        g1[3] = 0x3000 << 16;                         // tile_dim0   = 12288
        g1[4] = 1;                                    // tile_dim1   = 1
        g1[5] = 0x3000;                               // tensor_dim0_stride = 12288
        g1[6] = 0x3000 << 16;                         // tensor_dim1_stride lo
        g1[7] = 0;
        __builtin_amdgcn_tensor_load_to_lds(g0, g1, g2, g3, g4, 0);
    }

    // --- Stage x tile (+causal halo) as f16, vectorized float4 -> 4x f16 ---
    // One row per 32 lanes: each row = 32 lanes * float4 = 512B coalesced.
    // Row index (hence the t>=0 pad check) is uniform per wave.
    {
        int colb = lane * 4;
        for (int row = (tid >> 5); row < XROWS; row += 8) {
            int t = t0 - HALO + row;
            v4h hv = {0, 0, 0, 0};
            if (t >= 0) {
                const float4 v =
                    *(const float4*)&x[(((size_t)bb * TLEN + t) * NF + f) * CI + colb];
                hv[0] = (_Float16)v.x; hv[1] = (_Float16)v.y;
                hv[2] = (_Float16)v.z; hv[3] = (_Float16)v.w;
            }
            *(v4h*)&xs[row * PADROW + colb] = hv;
        }
    }

    if (wv == 0) __builtin_amdgcn_s_wait_tensorcnt(0);
    __syncthreads();

    int mrow = lane & 15;                 // A-matrix row within 16
    int grp8 = (lane & 16) >> 1;          // 0 (lanes<16) or 8 (lanes>=16)

    v8f acc[8] = {};                      // 8 n-tiles of 16 cols, f32 accum

    union Hv { v16h v; v8h h[2]; };

    #pragma unroll
    for (int kc = 0; kc < NKC; ++kc) {
        int k  = kc >> 2;                 // tap index
        int ib = (kc & 3) * 32;           // ci base within tap
        // A fragment, 16-bit A 16x32 layout:
        //   lanes<16:  halves 0-7 -> K ib+0..7,  halves 8-15 -> K ib+16..23
        //   lanes>=16: halves 0-7 -> K ib+8..15, halves 8-15 -> K ib+24..31
        int tl = wv * 16 + mrow + k * DILA;
        const _Float16* ap = &xs[tl * PADROW + ib + grp8];
        Hv af;
        af.h[0] = *(const v8h*)(ap);
        af.h[1] = *(const v8h*)(ap + 16);

        #pragma unroll
        for (int nt = 0; nt < 8; ++nt) {
            const _Float16* bp = &wlds[(kc * 8 + nt) * 512 + lane * 8];
            Hv bf;
            bf.h[0] = *(const v8h*)(bp);        // chunk 0: halves 0-7
            bf.h[1] = *(const v8h*)(bp + 256);  // chunk 1: halves 8-15
            acc[nt] = __builtin_amdgcn_wmma_f32_16x16x32_f16(
                false, af.v, false, bf.v, (short)0, acc[nt], false, false);
        }
    }

    // Epilogue: C/D 16x16 f32 layout: VGPR r -> M = r (lanes<16) or r+8.
    int trow = t0 + wv * 16 + ((lane & 16) >> 1);
    int ncol = lane & 15;
    #pragma unroll
    for (int nt = 0; nt < 8; ++nt) {
        int o = nt * 16 + ncol;
        float bv = bias[f * CO + o];
        #pragma unroll
        for (int r = 0; r < 8; ++r) {
            int t = trow + r;
            __builtin_nontemporal_store(
                acc[nt][r] + bv,
                &out[(((size_t)bb * TLEN + t) * NF + f) * CO + o]);
        }
    }
}

extern "C" void kernel_launch(void* const* d_in, const int* in_sizes, int n_in,
                              void* d_out, int out_size, void* d_ws, size_t ws_size,
                              hipStream_t stream) {
    const float* x = (const float*)d_in[0];   // [16][512][32][128]
    const float* w = (const float*)d_in[1];   // [32][128][128][3]
    const float* b = (const float*)d_in[2];   // [32][128]
    float* out = (float*)d_out;               // [16][512][32][128]
    _Float16* wsw = (_Float16*)d_ws;          // 3 MB of swizzled f16 weights

    int nsw = NF * WSWZ_PER_F;                // 1,572,864 elements
    swizzle_w_kernel<<<nsw / 256, 256, 0, stream>>>(w, wsw);
    conv_wmma_kernel<<<NF * 64, 256, SMEM_BYTES, stream>>>(x, wsw, b, out);
}
